// GraphTower_83846351553154
// MI455X (gfx1250) — matrix operand order repaired and checked by reading, not compiled, not run on hardware
//
#include <hip/hip_runtime.h>
#include <hip/hip_bf16.h>

// ---------------------------------------------------------------------------
// SchNet-style GraphTower, fused per-molecule for MI455X (gfx1250, wave32).
//  * All GEMMs via v_wmma_f32_16x16x32_f16 (f16 in, f32 accumulate).
//  * Weights pre-converted once to f16, transposed [N][K] so a B-fragment is
//    two 8-half contiguous runs per lane (two b128 loads, L2-resident: ~0.9 MB total).
//  * One workgroup (8 wave32) per molecule; h/x1/agg/distances live in LDS.
//  * Edge filter network: RBF A-fragments computed in-register (v_exp_f32),
//    scatter-add reduction via ds_add_f32 (one atomic per lane per tile).
//  * Tile loops kept rolled (#pragma unroll 1) to keep VGPR usage in the
//    ~130 range: round 1 fully unrolled them and spilled to scratch (>400
//    VGPRs, s_set_vgpr_msb + Folded Reloads), destroying occupancy.
// ---------------------------------------------------------------------------

typedef __attribute__((ext_vector_type(16))) _Float16 v16h;
typedef __attribute__((ext_vector_type(8)))  _Float16 v8h;
typedef __attribute__((ext_vector_type(8)))  float    v8f;

#define GMOL 1024
#define NATOM 30
#define AP 32          // atoms padded
#define HDIM 128
#define FDIM 128
#define NGAUSS 50
#define NLAYER 6
#define ODIM 128

// f16 transposed weight layout inside d_ws (offsets in halfs)
#define OFF_W1   0                         // [6][128][64]
#define OFF_W2   (OFF_W1  + 6*128*64)      // [6][128][128]
#define OFF_CF1  (OFF_W2  + 6*128*128)
#define OFF_CF2  (OFF_CF1 + 6*128*128)
#define OFF_INT  (OFF_CF2 + 6*128*128)
#define OFF_O1   (OFF_INT + 6*128*128)     // [64][128]
#define OFF_O2   (OFF_O1  + 64*128)        // [128][64]

struct Smem {
  float    posS[AP][3];
  float    dmat[AP][AP];     // pair distances
  float    cmat[AP][AP];     // cosine cutoff, 0 on self/pad edges
  float    hS[AP][HDIM];     // node features (residual stream)
  float    x1S[AP][FDIM];    // h @ cf_lin1
  float    aggS[AP][FDIM];   // scatter-add target
  _Float16 tS[128][128];     // f16 staging (t1 tiles / intermediate acts)
  _Float16 xhS[AP][HDIM];    // f16 copy of node matrix for A-fragments
};

__device__ __forceinline__ float ssp_f(float x) {
  return __logf(1.0f + __expf(x)) - 0.69314718055994531f;
}

__device__ __forceinline__ v8f wmma_f16(v16h a, v16h b, v8f c) {
  return __builtin_amdgcn_wmma_f32_16x16x32_f16(
      /*neg_a=*/false, a, /*neg_b=*/false, b,
      /*c_mod=*/(short)0, c, /*reuse_a=*/false, /*reuse_b=*/false);
}

// A fragment (16x32 f16, row-major source, leading dim `ld` halfs).
// Lanes 0-15: row m0+lane, K = k0+{0..7, 16..23}; lanes 16-31: +8 / +24.
__device__ __forceinline__ v16h frag_a(const _Float16* A, int ld, int m0, int k0) {
  const int lane = threadIdx.x & 31;
  const int row  = m0 + (lane & 15);
  const int kb   = k0 + ((lane >> 4) << 3);
  const v8h lo = *(const v8h*)(A + row * ld + kb);
  const v8h hi = *(const v8h*)(A + row * ld + kb + 16);
  v16h r;
#pragma unroll
  for (int t = 0; t < 8; ++t) { r[t] = lo[t]; r[t + 8] = hi[t]; }
  return r;
}

// B fragment (32x16 f16) from transposed weights Bt[n][k] (leading dim kld).
// Lanes 0-15: col n0+lane, K = k0+{0..7, 16..23}; lanes 16-31: +8 / +24.
__device__ __forceinline__ v16h frag_b(const _Float16* Bt, int kld, int n0, int k0) {
  const int lane = threadIdx.x & 31;
  const int col  = n0 + (lane & 15);
  const int kb   = k0 + ((lane >> 4) << 3);
  const v8h lo = *(const v8h*)(Bt + col * kld + kb);
  const v8h hi = *(const v8h*)(Bt + col * kld + kb + 16);
  v16h r;
#pragma unroll
  for (int t = 0; t < 8; ++t) { r[t] = lo[t]; r[t + 8] = hi[t]; }
  return r;
}

// ---------------------------------------------------------------------------
// Kernel 0: convert all weights to f16, transposed [fo][fi] (K padded for w1).
// ---------------------------------------------------------------------------
__device__ __forceinline__ void transpose128(const float* src, _Float16* dst, int tid) {
  for (int idx = tid; idx < 128 * 128; idx += 256) {
    int n = idx >> 7, k = idx & 127;
    dst[idx] = (_Float16)src[k * 128 + n];
  }
}

__global__ __launch_bounds__(256) void prep_weights(
    const float* w1, const float* w2, const float* cf1, const float* cf2,
    const float* intw, const float* o1, const float* o2, _Float16* ws) {
  const int b = blockIdx.x;      // 0..31
  const int tid = threadIdx.x;
  if (b < 6) {                   // mlp_w1: [50][128] -> [128][64] pad K
    const float* src = w1 + b * NGAUSS * FDIM;
    _Float16* dst = ws + OFF_W1 + b * 128 * 64;
    for (int idx = tid; idx < 128 * 64; idx += 256) {
      int n = idx >> 6, k = idx & 63;
      dst[idx] = (_Float16)((k < NGAUSS) ? src[k * FDIM + n] : 0.0f);
    }
  } else if (b < 12) {
    transpose128(w2  + (b - 6)  * 128 * 128, ws + OFF_W2  + (b - 6)  * 128 * 128, tid);
  } else if (b < 18) {
    transpose128(cf1 + (b - 12) * 128 * 128, ws + OFF_CF1 + (b - 12) * 128 * 128, tid);
  } else if (b < 24) {
    transpose128(cf2 + (b - 18) * 128 * 128, ws + OFF_CF2 + (b - 18) * 128 * 128, tid);
  } else if (b < 30) {
    transpose128(intw + (b - 24) * 128 * 128, ws + OFF_INT + (b - 24) * 128 * 128, tid);
  } else if (b == 30) {          // out_w1: [128][64] -> [64][128]
    _Float16* dst = ws + OFF_O1;
    for (int idx = tid; idx < 64 * 128; idx += 256) {
      int n = idx >> 7, k = idx & 127;
      dst[idx] = (_Float16)o1[k * 64 + n];
    }
  } else {                       // out_w2: [64][128] -> [128][64]
    _Float16* dst = ws + OFF_O2;
    for (int idx = tid; idx < 128 * 64; idx += 256) {
      int n = idx >> 6, k = idx & 63;
      dst[idx] = (_Float16)o2[k * 128 + n];
    }
  }
}

// ---------------------------------------------------------------------------
// Kernel 1: fully fused SchNet, one workgroup per molecule.
// ---------------------------------------------------------------------------
__global__ __launch_bounds__(256) void schnet_fused(
    const int* __restrict__ z, const float* __restrict__ pos,
    const float* __restrict__ emb,
    const float* __restrict__ mb1, const float* __restrict__ mb2,
    const float* __restrict__ cf2b, const float* __restrict__ intb,
    const float* __restrict__ ob1, const float* __restrict__ ob2,
    const _Float16* __restrict__ w1T, const _Float16* __restrict__ w2T,
    const _Float16* __restrict__ cf1T, const _Float16* __restrict__ cf2T,
    const _Float16* __restrict__ intT, const _Float16* __restrict__ o1T,
    const _Float16* __restrict__ o2T, float* __restrict__ out) {
  extern __shared__ char smem_raw[];
  Smem& S = *reinterpret_cast<Smem*>(smem_raw);

  const int g    = blockIdx.x;
  const int tid  = threadIdx.x;
  const int wave = tid >> 5;
  const int lane = tid & 31;
  const int rb   = (lane < 16) ? 0 : 8;   // C/D fragment row-half offset

  const float step  = 10.0f / 49.0f;                  // gaussian spacing
  const float gcoef = -0.5f / (step * step);

  // --- stage positions + embedding into LDS --------------------------------
  for (int idx = tid; idx < AP * 3; idx += 256) {
    int a = idx / 3, c = idx % 3;
    S.posS[a][c] = (a < NATOM) ? pos[(g * NATOM + a) * 3 + c] : 0.0f;
  }
  for (int idx = tid; idx < AP * HDIM; idx += 256) {
    int a = idx >> 7, f = idx & 127;
    float v = 0.0f;
    if (a < NATOM) v = emb[z[g * NATOM + a] * HDIM + f];
    S.hS[a][f] = v;
  }
  __syncthreads();

  // --- pairwise distances + masked cosine cutoff (layer-invariant) ---------
  for (int idx = tid; idx < AP * AP; idx += 256) {
    int i = idx >> 5, j = idx & 31;
    float dx = S.posS[i][0] - S.posS[j][0];
    float dy = S.posS[i][1] - S.posS[j][1];
    float dz = S.posS[i][2] - S.posS[j][2];
    float d  = __builtin_sqrtf(dx * dx + dy * dy + dz * dz);
    S.dmat[i][j] = d;
    float c = 0.0f;
    if (i < NATOM && j < NATOM && i != j)
      c = 0.5f * (__cosf(d * 0.31415926535f) + 1.0f);
    S.cmat[i][j] = c;
  }
  __syncthreads();

  // ========================== interaction layers ===========================
#pragma unroll 1
  for (int l = 0; l < NLAYER; ++l) {
    const _Float16* w1l  = w1T  + l * 128 * 64;
    const _Float16* w2l  = w2T  + l * 128 * 128;
    const _Float16* cf1l = cf1T + l * 128 * 128;
    const _Float16* cf2l = cf2T + l * 128 * 128;
    const _Float16* intl = intT + l * 128 * 128;
    const float* b1l  = mb1  + l * 128;
    const float* b2l  = mb2  + l * 128;
    const float* c2bl = cf2b + l * 128;
    const float* ibl  = intb + l * 128;
    __builtin_prefetch(w2l, 0, 1);
    __builtin_prefetch(cf1l, 0, 1);

    // ---- x1 = h @ cf_lin1 (no bias): 2x8 tiles of 16x16, K=128 ----
    for (int idx = tid; idx < AP * HDIM; idx += 256)
      S.xhS[idx >> 7][idx & 127] = (_Float16)S.hS[idx >> 7][idx & 127];
    __syncthreads();
#pragma unroll 1
    for (int t = wave; t < 16; t += 8) {
      int mt = t >> 3, nt = t & 7;
      v8f acc = {};
#pragma unroll
      for (int k = 0; k < 128; k += 32)
        acc = wmma_f16(frag_a(&S.xhS[0][0], 128, mt * 16, k),
                       frag_b(cf1l, 128, nt * 16, k), acc);
      int col = nt * 16 + (lane & 15);
#pragma unroll
      for (int r = 0; r < 8; ++r)
        S.x1S[mt * 16 + rb + r][col] = acc[r];
    }
    for (int idx = tid; idx < AP * FDIM; idx += 256)
      S.aggS[idx >> 7][idx & 127] = 0.0f;
    __syncthreads();

    // ---- edge filter network + message aggregation ----
    // pair blocks: 8 dst (one per wave) x 16 src = 128 pairs, 8 blocks total
#pragma unroll 1
    for (int pb = 0; pb < 8; ++pb) {
      const int jb = pb >> 1, ib = pb & 1;
      const int j  = jb * 8 + wave;          // this wave's destination atom

      // RBF A-fragments (16 pairs x 64 gaussians) computed in-register
      v16h ra[2];
      {
        int i = ib * 16 + (lane & 15);
        float d = S.dmat[i][j];
#pragma unroll
        for (int kk = 0; kk < 2; ++kk) {
          int kb = kk * 32 + ((lane >> 4) << 3);
          v16h r;
#pragma unroll
          for (int t = 0; t < 16; ++t) {
            int k = kb + ((t >> 3) << 4) + (t & 7);
            float v = 0.0f;
            if (k < NGAUSS) { float dd = d - (float)k * step; v = __expf(gcoef * dd * dd); }
            r[t] = (_Float16)v;
          }
          ra[kk] = r;
        }
      }
      // t1 = ssp(rbf @ w1 + b1) -> LDS rows owned by this wave (no barrier)
#pragma unroll 1
      for (int nt = 0; nt < 8; ++nt) {
        v8f acc = {};
        acc = wmma_f16(ra[0], frag_b(w1l, 64, nt * 16, 0),  acc);
        acc = wmma_f16(ra[1], frag_b(w1l, 64, nt * 16, 32), acc);
        int col = nt * 16 + (lane & 15);
        float bb = b1l[col];
#pragma unroll
        for (int r = 0; r < 8; ++r)
          S.tS[wave * 16 + rb + r][col] = (_Float16)ssp_f(acc[r] + bb);
      }
      // W = t1 @ w2 + b2; msg = W * C * x1[src]; column-sum over 16 srcs
      v16h ta[4];
#pragma unroll
      for (int k = 0; k < 4; ++k)
        ta[k] = frag_a(&S.tS[0][0], 128, wave * 16, k * 32);
#pragma unroll 1
      for (int nt = 0; nt < 8; ++nt) {
        v8f acc = {};
#pragma unroll
        for (int k = 0; k < 4; ++k)
          acc = wmma_f16(ta[k], frag_b(w2l, 128, nt * 16, k * 32), acc);
        int col = nt * 16 + (lane & 15);
        float bb = b2l[col];
        float s = 0.0f;
#pragma unroll
        for (int r = 0; r < 8; ++r) {
          int i = ib * 16 + rb + r;
          s += (acc[r] + bb) * S.cmat[i][j] * S.x1S[i][col];
        }
        atomicAdd(&S.aggS[j][col], s);   // ds_add_f32
      }
    }
    __syncthreads();

    // ---- x = ssp(agg @ cf_lin2 + b) ----
    for (int idx = tid; idx < AP * FDIM; idx += 256)
      S.xhS[idx >> 7][idx & 127] = (_Float16)S.aggS[idx >> 7][idx & 127];
    __syncthreads();
#pragma unroll 1
    for (int t = wave; t < 16; t += 8) {
      int mt = t >> 3, nt = t & 7;
      v8f acc = {};
#pragma unroll
      for (int k = 0; k < 128; k += 32)
        acc = wmma_f16(frag_a(&S.xhS[0][0], 128, mt * 16, k),
                       frag_b(cf2l, 128, nt * 16, k), acc);
      int col = nt * 16 + (lane & 15);
      float bb = c2bl[col];
#pragma unroll
      for (int r = 0; r < 8; ++r)
        S.tS[mt * 16 + rb + r][col] = (_Float16)ssp_f(acc[r] + bb);
    }
    __syncthreads();

    // ---- h += x @ int_w + int_b (residual) ----
#pragma unroll 1
    for (int t = wave; t < 16; t += 8) {
      int mt = t >> 3, nt = t & 7;
      v8f acc = {};
#pragma unroll
      for (int k = 0; k < 128; k += 32)
        acc = wmma_f16(frag_a(&S.tS[0][0], 128, mt * 16, k),
                       frag_b(intl, 128, nt * 16, k), acc);
      int col = nt * 16 + (lane & 15);
      float bb = ibl[col];
#pragma unroll
      for (int r = 0; r < 8; ++r)
        S.hS[mt * 16 + rb + r][col] += acc[r] + bb;
    }
    __syncthreads();
  }

  // ============================ output head ================================
  for (int idx = tid; idx < AP * HDIM; idx += 256)
    S.xhS[idx >> 7][idx & 127] = (_Float16)S.hS[idx >> 7][idx & 127];
  __syncthreads();

  // ho = ssp(h @ out_w1 + b1): [32][64], 2x4 tiles, one per wave
  {
    int mt = wave >> 2, nt = wave & 3;
    v8f acc = {};
#pragma unroll
    for (int k = 0; k < 128; k += 32)
      acc = wmma_f16(frag_a(&S.xhS[0][0], 128, mt * 16, k),
                     frag_b(o1T, 128, nt * 16, k), acc);
    int col = nt * 16 + (lane & 15);
    float bb = ob1[col];
#pragma unroll
    for (int r = 0; r < 8; ++r)
      S.tS[0][(mt * 16 + rb + r) * 64 + col] = (_Float16)ssp_f(acc[r] + bb);
  }
  if (tid < ODIM) S.aggS[0][tid] = 0.0f;   // per-graph accumulator
  __syncthreads();

  // out = sum_atoms (ho @ out_w2 + b2): [32][128], K=64
#pragma unroll 1
  for (int t = wave; t < 16; t += 8) {
    int mt = t >> 3, nt = t & 7;
    v8f acc = {};
#pragma unroll
    for (int k = 0; k < 64; k += 32)
      acc = wmma_f16(frag_a(&S.tS[0][0], 64, mt * 16, k),
                     frag_b(o2T, 64, nt * 16, k), acc);
    int col = nt * 16 + (lane & 15);
    float bb = ob2[col];
    float s = 0.0f;
#pragma unroll
    for (int r = 0; r < 8; ++r) {
      int atom = mt * 16 + rb + r;
      if (atom < NATOM) s += acc[r] + bb;
    }
    atomicAdd(&S.aggS[0][col], s);
  }
  __syncthreads();
  if (tid < ODIM) out[g * ODIM + tid] = S.aggS[0][tid];
}

// ---------------------------------------------------------------------------
extern "C" void kernel_launch(void* const* d_in, const int* in_sizes, int n_in,
                              void* d_out, int out_size, void* d_ws, size_t ws_size,
                              hipStream_t stream) {
  (void)in_sizes; (void)n_in; (void)out_size; (void)ws_size;
  const int*   z    = (const int*)  d_in[0];
  const float* pos  = (const float*)d_in[1];
  // d_in[2]=batch, d_in[3]=edge_index: implicit in the complete-graph layout
  const float* emb  = (const float*)d_in[4];
  const float* w1   = (const float*)d_in[5];
  const float* b1   = (const float*)d_in[6];
  const float* w2   = (const float*)d_in[7];
  const float* b2   = (const float*)d_in[8];
  const float* cf1  = (const float*)d_in[9];
  const float* cf2  = (const float*)d_in[10];
  const float* cf2b = (const float*)d_in[11];
  const float* intw = (const float*)d_in[12];
  const float* intb = (const float*)d_in[13];
  const float* o1   = (const float*)d_in[14];
  const float* ob1  = (const float*)d_in[15];
  const float* o2   = (const float*)d_in[16];
  const float* ob2  = (const float*)d_in[17];
  _Float16* ws = (_Float16*)d_ws;

  (void)hipFuncSetAttribute((const void*)schnet_fused,
                            hipFuncAttributeMaxDynamicSharedMemorySize,
                            (int)sizeof(Smem));

  prep_weights<<<32, 256, 0, stream>>>(w1, w2, cf1, cf2, intw, o1, o2, ws);
  schnet_fused<<<GMOL, 256, sizeof(Smem), stream>>>(
      z, pos, emb, b1, b2, cf2b, intb, ob1, ob2,
      ws + OFF_W1, ws + OFF_W2, ws + OFF_CF1, ws + OFF_CF2, ws + OFF_INT,
      ws + OFF_O1, ws + OFF_O2, (float*)d_out);
}
